// Graph_84920093377318
// MI455X (gfx1250) — matrix operand order repaired and checked by reading, not compiled, not run on hardware
//
#include <hip/hip_runtime.h>
#include <hip/hip_bf16.h>
#include <math.h>

// Problem constants
#define BSZ   32
#define PPART 3
#define CIN   64
#define EMB   64
#define TT    300
#define NN    25
#define NP    32                 // padded node dim
#define NCHUNK 15                // 15 chunks x 20 frames = 300
#define FR_PER_IT 4              // frames staged per inner iteration
#define IT_PER_CHUNK 5           // 5 * 4 = 20 frames per block
#define XLD   264                // padded LDS row stride (halfs): 528B = 4 banks mod 64
#define WLD   72                 // padded W_p row stride (halfs): 144B = 36 banks mod 64

typedef _Float16 v16h __attribute__((ext_vector_type(16)));
typedef _Float16 v8h  __attribute__((ext_vector_type(8)));
typedef float    v8f  __attribute__((ext_vector_type(8)));

// ---- WMMA f16 fragment loaders (wave32 layouts, CDNA5 ISA 7.12.2) --------
// Per-lane data is contiguous in LDS (transposed layouts), so each fragment
// is two ds_load_b128 + a register concat.

__device__ inline v16h ldA16(const _Float16* p) {   // p = row + k0 + hs*8
  v8h lo = *(const v8h*)p;
  v8h hi = *(const v8h*)(p + 16);
  return __builtin_shufflevector(lo, hi, 0, 1, 2, 3, 4, 5, 6, 7,
                                 8, 9, 10, 11, 12, 13, 14, 15);
}

__device__ inline v16h ldB16(const _Float16* p) {   // p = row + k0 + hs*8
  v8h lo = *(const v8h*)p;
  v8h hi = *(const v8h*)(p + 16);
  return __builtin_shufflevector(lo, hi, 0, 1, 2, 3, 4, 5, 6, 7,
                                 8, 9, 10, 11, 12, 13, 14, 15);
}

__device__ inline v8f wmma_f16(v16h a, v16h b, v8f c) {
  return __builtin_amdgcn_wmma_f32_16x16x32_f16(false, a, false, b, (short)0, c,
                                                false, false);
}

// ------------------- prep: W_p = phi_w^T theta_w, bias vectors -------------
__global__ void __launch_bounds__(256)
prep_kernel(const float* __restrict__ phi_w, const float* __restrict__ phi_b,
            const float* __restrict__ theta_w, const float* __restrict__ theta_b,
            _Float16* __restrict__ Wp_ws, float* __restrict__ uv_ws) {
  const int p = blockIdx.x, tid = threadIdx.x;
  const float* pw = phi_w   + p * EMB * CIN;
  const float* tw = theta_w + p * EMB * CIN;
  const float* pb = phi_b   + p * EMB;
  const float* tb = theta_b + p * EMB;
  for (int idx = tid; idx < CIN * CIN; idx += 256) {
    const int c = idx >> 6, c2 = idx & 63;
    float s = 0.f;
    for (int e = 0; e < EMB; ++e) s += pw[e * CIN + c] * tw[e * CIN + c2];
    Wp_ws[p * CIN * CIN + idx] = (_Float16)s;
  }
  if (tid < CIN) {
    float su = 0.f, sv = 0.f;
    for (int e = 0; e < EMB; ++e) {
      su += pb[e] * tw[e * CIN + tid];     // u_p . Sx[:,m]  (theta side)
      sv += tb[e] * pw[e * CIN + tid];     // v_p . Sx[:,n]  (phi side)
    }
    uv_ws[p * CIN + tid] = su;
    uv_ws[PPART * CIN + p * CIN + tid] = sv;
  }
  if (tid == CIN) {
    float ss = 0.f;
    for (int e = 0; e < EMB; ++e) ss += pb[e] * tb[e];
    uv_ws[2 * PPART * CIN + p] = ss;
  }
}

// ------------------- Sx[b,c,n] = sum_t x[b,c,t,n] --------------------------
__global__ void __launch_bounds__(32)
sumx_kernel(const float* __restrict__ x, float* __restrict__ Sx) {
  const int bc = blockIdx.x;                 // b*CIN + c
  const int n = threadIdx.x;
  if (n >= NN) return;
  const float* base = x + (size_t)bc * TT * NN;
  float s = 0.f;
  for (int t = 0; t < TT; ++t) s += base[t * NN + n];   // 25-lane coalesced
  Sx[bc * NN + n] = s;
}

// ------------------- main WMMA kernel: all 3 partitions share staged x -----
__global__ void __launch_bounds__(256)
scores_kernel(const float* __restrict__ x, const _Float16* __restrict__ Wp_all,
              float* __restrict__ partial_ws) {
  const int b = blockIdx.x, chunk = blockIdx.y;
  const int tid = threadIdx.x;
  const int lane = tid & 31, wave = tid >> 5;
  const int hs = (lane >> 4) & 1;            // lane half-select
  const int lnid = lane & 15;
  const int r8 = hs * 8;

  __shared__ __align__(16) _Float16 lds_w[PPART * CIN * WLD];  // 27648 B
  __shared__ __align__(16) unsigned long long raw[4224];       // 33792 B, dual-use
  _Float16* lds_xT = (_Float16*)raw;           // [n][K=(f,c)]  32 x XLD halfs
  _Float16* lds_uT = lds_xT + NP * XLD;        // [m][K=(f,c)]  32 x XLD halfs
  float*    parts  = (float*)raw;              // 8 waves * 1024 f32 (reuse)

  // Stage all three W_p into padded LDS; zero x staging (pad rows stay zero).
  for (int i = tid; i < PPART * CIN * CIN; i += 256) {
    const int pp = i >> 12, r = i & 4095;
    lds_w[pp * (CIN * WLD) + (r >> 6) * WLD + (r & 63)] = Wp_all[i];
  }
  for (int i = tid; i < (NP * XLD) / 4; i += 256) ((unsigned long long*)raw)[i] = 0ull;
  __syncthreads();

  const int mtW = wave >> 1;                 // fixed W_p row-tile per wave
  v8f acc[PPART][2][2] = {};                 // 12 x v8f accumulators

  const int t0 = chunk * (FR_PER_IT * IT_PER_CHUNK);
  const float* xb = x + (size_t)b * CIN * TT * NN;

  for (int it = 0; it < IT_PER_CHUNK; ++it) {
    const int tf = t0 + it * FR_PER_IT;
    // stage 4 frames of x -> f16, node-major [n][f*64+c] (global reads coalesced)
    for (int i = tid; i < FR_PER_IT * CIN * NN; i += 256) {
      const int f = i / (CIN * NN), r = i - f * (CIN * NN);
      const int c = r / NN, n = r - c * NN;
      lds_xT[n * XLD + f * CIN + c] =
          (_Float16)xb[((size_t)c * TT + (tf + f)) * NN + n];
    }
    __syncthreads();
    // x^T A-fragments for GEMM2: shared by all 3 partitions this iteration
    const int k0 = wave * 32;
    const v16h aT0 = ldA16(lds_xT + lnid * XLD + k0 + r8);
    const v16h aT1 = ldA16(lds_xT + (16 + lnid) * XLD + k0 + r8);

#pragma unroll
    for (int pp = 0; pp < PPART; ++pp) {
      // GEMM1: u = W_p(64x64) @ x(64 x 4*32); each wave: fixed mt, 4 col tiles
      const _Float16* wrow = lds_w + pp * (CIN * WLD) + (mtW * 16 + lnid) * WLD;
      const v16h aW0 = ldA16(wrow + 0 + r8);   // K(c') 0..31
      const v16h aW1 = ldA16(wrow + 32 + r8);  // K(c') 32..63
#pragma unroll
      for (int i = 0; i < 4; ++i) {
        const int nt = (wave & 1) * 4 + i;     // column tile 0..7
        const int f = nt >> 1, n0 = (nt & 1) * 16;
        const _Float16* xrow = lds_xT + (n0 + lnid) * XLD + f * CIN;
        v16h b0 = ldB16(xrow + 0 + r8);
        v16h b1 = ldB16(xrow + 32 + r8);
        v8f d = {};
        d = wmma_f16(aW0, b0, d);
        d = wmma_f16(aW1, b1, d);
        v8h uh;                                // D tile -> u^T, one b128 store
#pragma unroll
        for (int j = 0; j < 8; ++j) uh[j] = (_Float16)d[j];
        *(v8h*)(lds_uT + (n0 + lnid) * XLD + f * CIN + mtW * 16 + r8) = uh;
      }
      __syncthreads();
      // GEMM2: scores_p(32x32) += x^T(32 x 256) @ u_p(256 x 32); K-step = wave
      {
        v16h bu0 = ldB16(lds_uT + lnid * XLD + k0 + r8);
        v16h bu1 = ldB16(lds_uT + (16 + lnid) * XLD + k0 + r8);
        acc[pp][0][0] = wmma_f16(aT0, bu0, acc[pp][0][0]);
        acc[pp][0][1] = wmma_f16(aT0, bu1, acc[pp][0][1]);
        acc[pp][1][0] = wmma_f16(aT1, bu0, acc[pp][1][0]);
        acc[pp][1][1] = wmma_f16(aT1, bu1, acc[pp][1][1]);
      }
      __syncthreads();   // u buffer free for next partition / next staging
    }
  }

  // Deterministic cross-wave reduction (no atomics), one partition at a time.
  const int r0 = hs ? 8 : 0;
#pragma unroll
  for (int pp = 0; pp < PPART; ++pp) {
    __syncthreads();
    float* mp = parts + wave * (NP * NP);
#pragma unroll
    for (int j = 0; j < 8; ++j) {
      mp[(r0 + j) * NP + lnid]           = acc[pp][0][0][j];
      mp[(r0 + j) * NP + 16 + lnid]      = acc[pp][0][1][j];
      mp[(16 + r0 + j) * NP + lnid]      = acc[pp][1][0][j];
      mp[(16 + r0 + j) * NP + 16 + lnid] = acc[pp][1][1][j];
    }
    __syncthreads();
    float* outp = partial_ws +
                  ((size_t)((b * PPART + pp) * NCHUNK) + chunk) * (NP * NP);
    for (int i = tid; i < NP * NP; i += 256) {
      float s = 0.f;
#pragma unroll
      for (int w = 0; w < 8; ++w) s += parts[w * (NP * NP) + i];
      outp[i] = s;
    }
  }
}

// ------------------- finalize: bias + scale + softmax + combine ------------
__global__ void __launch_bounds__(32)
finalize_kernel(const float* __restrict__ partial_ws, const float* __restrict__ Sx,
                const float* __restrict__ uv, const float* __restrict__ A,
                const float* __restrict__ Bm, const float* __restrict__ ei,
                float* __restrict__ out) {
  const int b = blockIdx.x, p = blockIdx.y;
  const int tid = threadIdx.x;
  __shared__ float dotv[NP];
  const float* u = uv + p * CIN;
  const float* v = uv + PPART * CIN + p * CIN;
  const float sp = uv[2 * PPART * CIN + p];
  const float* Sxb = Sx + (size_t)b * CIN * NN;
  if (tid < NN) {
    float dv = 0.f;
    for (int c = 0; c < CIN; ++c) dv += v[c] * Sxb[c * NN + tid];
    dotv[tid] = dv;
  }
  __syncthreads();
  if (tid >= NN) return;
  const int m = tid;
  float du = 0.f;
  for (int c = 0; c < CIN; ++c) du += u[c] * Sxb[c * NN + m];
  const float invScale = 1.0f / ((float)EMB * (float)TT);
  const float bias = du + (float)TT * sp;
  const float* qbase = partial_ws + (size_t)((b * PPART + p) * NCHUNK) * (NP * NP);
  float col[NN];
  float mx = -3.402823466e38f;
  for (int n = 0; n < NN; ++n) {
    float q = 0.f;
    for (int ch = 0; ch < NCHUNK; ++ch) q += qbase[ch * (NP * NP) + n * NP + m];
    const float s = (q + bias + dotv[n]) * invScale;
    col[n] = s;
    mx = fmaxf(mx, s);
  }
  float ssum = 0.f;
  for (int n = 0; n < NN; ++n) { col[n] = expf(col[n] - mx); ssum += col[n]; }
  const float inv = 1.0f / ssum;
  for (int n = 0; n < NN; ++n) {
    const float Cnm = col[n] * inv;
    const int aidx = (p * NN + n) * NN + m;
    out[((b * PPART + p) * NN + n) * NN + m] = (A[aidx] + Bm[aidx] + Cnm) * ei[aidx];
  }
}

extern "C" void kernel_launch(void* const* d_in, const int* in_sizes, int n_in,
                              void* d_out, int out_size, void* d_ws, size_t ws_size,
                              hipStream_t stream) {
  const float* x       = (const float*)d_in[0];  // (32,64,300,25)
  const float* A       = (const float*)d_in[1];  // (3,25,25)
  const float* Bm      = (const float*)d_in[2];  // (3,25,25)
  const float* ei      = (const float*)d_in[3];  // (3,25,25)
  const float* phi_w   = (const float*)d_in[4];  // (3,64,64)
  const float* phi_b   = (const float*)d_in[5];  // (3,64)
  const float* theta_w = (const float*)d_in[6];  // (3,64,64)
  const float* theta_b = (const float*)d_in[7];  // (3,64)
  float* out = (float*)d_out;                    // (32,3,25,25)

  // Workspace layout (≈6.1 MB, deterministic fixed-order reductions)
  float* ws_partial = (float*)d_ws;                                   // 32*3*15*1024
  float* ws_Sx = ws_partial + (size_t)BSZ * PPART * NCHUNK * NP * NP; // 32*64*25
  float* ws_uv = ws_Sx + (size_t)BSZ * CIN * NN;                      // 512 floats
  _Float16* ws_Wp = (_Float16*)(ws_uv + 512);                         // 3*64*64 f16

  prep_kernel<<<PPART, 256, 0, stream>>>(phi_w, phi_b, theta_w, theta_b, ws_Wp, ws_uv);
  sumx_kernel<<<BSZ * CIN, 32, 0, stream>>>(x, ws_Sx);
  scores_kernel<<<dim3(BSZ, NCHUNK), 256, 0, stream>>>(x, ws_Wp, ws_partial);
  finalize_kernel<<<dim3(BSZ, PPART), 32, 0, stream>>>(ws_partial, ws_Sx, ws_uv,
                                                       A, Bm, ei, out);
  (void)in_sizes; (void)n_in; (void)out_size; (void)ws_size;
}
